// BiLSTMCRF_54176717472193
// MI455X (gfx1250) — matrix operand order repaired and checked by reading, not compiled, not run on hardware
//
#include <hip/hip_runtime.h>
#include <hip/hip_bf16.h>
#include <math.h>

// ---------------------------------------------------------------------------
// BiLSTM-CRF for gfx1250 (MI455X), wave32 + WMMA f32_16x16x32_f16.
//
// Pipeline (all on `stream`):
//   k_zero_out  : d_out[0] = 0
//   k_pack_ih   : w_ih_{f,b} (1024x350) -> f16 WMMA-B packed, K padded to 352
//   k_pack_hh   : w_hh_{f,b} (1024x256) -> f16 WMMA-B packed
//   k_embed     : word-emb gather + char-CNN(relu,maxpool) -> inp_f16 (16384x352)
//   k_xg        : WMMA GEMM: xg[dir] = inp @ w_ih[dir]^T + b[dir]  (16384x1024 f32)
//   k_lstm      : recurrent WMMA per 16-batch tile per direction; c in regs,
//                 h in LDS; handles fwd/bwd time indexing + masking
//   k_proj      : emissions = hcat @ proj_w^T + proj_b  (16384x17)
//   k_crf       : per-batch CRF NLL, atomicAdd into d_out
//
// WMMA operand packing (ISA 7.12.2, 16-bit A 16x32; B assumed mirrored):
//   A: lane L -> row m = L&15, half = L>>4; elements 0..7  = K[8*half .. +7],
//                                           elements 8..15 = K[16+8*half .. +7]
//   B: lane L -> col n = L&15, half = L>>4; same K order as A (0..7 = K[8*half .. +7], 8..15 = K[16+8*half .. +7])
//   C/D f32: lane L, reg r -> (m = r + 8*(L>>4), n = L&15)
// ---------------------------------------------------------------------------

typedef __attribute__((ext_vector_type(16))) _Float16 v16h;
typedef __attribute__((ext_vector_type(8)))  _Float16 v8h;
typedef __attribute__((ext_vector_type(8)))  float    v8f;

#define B_   128
#define S_   128
#define WL_  16
#define T_   17
#define WE_  300
#define CE_  50
#define CF_  50
#define K_   3
#define H_   256
#define D_   350
#define DP_  352            // D padded to multiple of 32
#define NTOK (B_*S_)        // 16384 tokens
#define G4H  (4*H_)         // 1024 gate outputs
#define NT_IH 64            // 1024/16 N-tiles
#define KK_IH 11            // 352/32 K-chunks
#define NT_HH 64
#define KK_HH 8             // 256/32 K-chunks

__device__ __forceinline__ float sigmoidf_(float x){ return 1.f/(1.f+__expf(-x)); }

// ---------------------------------------------------------------- zero output
__global__ void k_zero_out(float* out){
  if(blockIdx.x==0 && threadIdx.x==0) out[0]=0.f;
}

// ------------------------------------------------------------- weight packing
__global__ void k_pack_ih(const float* __restrict__ wf, const float* __restrict__ wb,
                          _Float16* __restrict__ pack){
  int idx = blockIdx.x*blockDim.x + threadIdx.x;
  const int total = 2*NT_IH*KK_IH*32*16;
  if(idx >= total) return;
  int e    =  idx        & 15;
  int lane = (idx >> 4)  & 31;
  int rest =  idx >> 9;                    // dir*NT*KK + nt*KK + kk
  int dir  = rest / (NT_IH*KK_IH);
  int rem  = rest % (NT_IH*KK_IH);
  int nt   = rem / KK_IH, kk = rem % KK_IH;
  int n = nt*16 + (lane & 15);
  int k = kk*32 + 8*(lane >> 4) + (e & 7) + 16*(e >> 3);
  const float* w = dir ? wb : wf;
  float v = (k < D_) ? w[n*D_ + k] : 0.f;
  pack[idx] = (_Float16)v;
}

__global__ void k_pack_hh(const float* __restrict__ wf, const float* __restrict__ wb,
                          _Float16* __restrict__ pack){
  int idx = blockIdx.x*blockDim.x + threadIdx.x;
  const int total = 2*NT_HH*KK_HH*32*16;
  if(idx >= total) return;
  int e    =  idx        & 15;
  int lane = (idx >> 4)  & 31;
  int rest =  idx >> 9;
  int dir  = rest / (NT_HH*KK_HH);
  int rem  = rest % (NT_HH*KK_HH);
  int nt   = rem / KK_HH, kk = rem % KK_HH;
  int n = nt*16 + (lane & 15);
  int k = kk*32 + 8*(lane >> 4) + (e & 7) + 16*(e >> 3);
  const float* w = dir ? wb : wf;
  pack[idx] = (_Float16)w[n*H_ + k];
}

// --------------------------------------------- embeddings + char CNN -> inp_f16
__global__ void k_embed(const int* __restrict__ words, const int* __restrict__ chars,
                        const float* __restrict__ word_emb, const float* __restrict__ char_emb,
                        const float* __restrict__ conv_w,  const float* __restrict__ conv_b,
                        _Float16* __restrict__ inp){
  int tok = blockIdx.x;
  int tid = threadIdx.x;
  __shared__ float ce[WL_][CE_];
  __shared__ float cfeat[CF_];
  __shared__ int   ch[WL_];
  if(tid < WL_) ch[tid] = chars[tok*WL_ + tid];
  __syncthreads();
  for(int i=tid; i<WL_*CE_; i+=blockDim.x){
    int t = i / CE_, e = i % CE_;
    ce[t][e] = char_emb[ch[t]*CE_ + e];
  }
  __syncthreads();
  if(tid < CF_){
    int cf = tid;
    float bias = conv_b[cf];
    float best = 0.f;                      // relu >= 0, so 0 is the identity for max(relu)
    for(int t=0; t<WL_; ++t){
      float acc = bias;
      #pragma unroll
      for(int k=0; k<K_; ++k){
        int tt = t + k - 1;                // SAME padding, K=3
        if(tt < 0 || tt >= WL_) continue;
        const float* wrow = conv_w + cf*CE_*K_ + k;
        for(int e=0; e<CE_; ++e) acc += ce[tt][e] * wrow[e*K_];
      }
      best = fmaxf(best, acc);
    }
    cfeat[cf] = best;
  }
  __syncthreads();
  int w = words[tok];
  for(int j=tid; j<DP_; j+=blockDim.x){
    float v = (j < WE_) ? word_emb[(size_t)w*WE_ + j]
            : (j < D_)  ? cfeat[j - WE_] : 0.f;
    inp[(size_t)tok*DP_ + j] = (_Float16)v;
  }
}

// -------------------------------- input projection GEMM: xg = inp @ w_ih^T + b
__global__ void k_xg(const _Float16* __restrict__ inp, const _Float16* __restrict__ pack_ih,
                     const float* __restrict__ b_f, const float* __restrict__ b_b,
                     float* __restrict__ xg){
  int wt   = blockIdx.x*8 + (threadIdx.x >> 5);   // wave-tile id, 131072 total
  int lane = threadIdx.x & 31;
  int dir  = wt >> 16;                            // 65536 tiles per direction
  int rem  = wt & 65535;
  int mt   = rem >> 6;                            // 1024 M-tiles
  int nt   = rem & 63;                            // 64  N-tiles
  int col  = lane & 15, half = lane >> 4;

  const _Float16* packd = pack_ih + (size_t)dir*NT_IH*KK_IH*512;
  const _Float16* arow  = inp + (size_t)(mt*16 + col)*DP_;
  v8f acc = {};
  #pragma unroll
  for(int kk=0; kk<KK_IH; ++kk){
    v8h lo = *(const v8h*)(arow + kk*32 +      8*half);
    v8h hi = *(const v8h*)(arow + kk*32 + 16 + 8*half);
    v16h a;
    #pragma unroll
    for(int e=0; e<8; ++e){ a[e] = lo[e]; a[e+8] = hi[e]; }
    v16h b = *(const v16h*)(packd + ((size_t)(nt*KK_IH + kk)*32 + lane)*16);
    acc = __builtin_amdgcn_wmma_f32_16x16x32_f16(false, a, false, b,
                                                 (short)0, acc, false, false);
  }
  int n = nt*16 + col;
  float bias = (dir ? b_b : b_f)[n];
  float* xgd = xg + (size_t)dir*NTOK*G4H;
  #pragma unroll
  for(int r=0; r<8; ++r){
    int m = mt*16 + r + 8*half;
    xgd[(size_t)m*G4H + n] = acc[r] + bias;
  }
}

// ------------------------------------------------- recurrent LSTM (fwd & bwd)
// grid (8,2): x = 16-row batch tile, y = direction. 512 threads = 16 waves.
// Wave w owns gate N-tiles {g*16+w : g=0..3} -> units u in [16w, 16w+16).
__global__ void k_lstm(const _Float16* __restrict__ pack_hh, const float* __restrict__ xg,
                       const int* __restrict__ lengths, float* __restrict__ hcat){
  int btile = blockIdx.x;
  int dir   = blockIdx.y;
  int tid   = threadIdx.x;
  int wv    = tid >> 5, lane = tid & 31;
  int col   = lane & 15, half = lane >> 4;
  int u     = wv*16 + col;

  __shared__ __align__(16) _Float16 hbuf[16*H_];     // h state, f16, 8 KB
  for(int i=tid; i<16*H_; i+=blockDim.x) hbuf[i] = (_Float16)0.f;

  const _Float16* packd = pack_hh + (size_t)dir*NT_HH*KK_HH*512;
  const float*    xgd   = xg + (size_t)dir*NTOK*G4H;

  int lens[8];
  #pragma unroll
  for(int r=0; r<8; ++r) lens[r] = lengths[btile*16 + r + 8*half];
  float creg[8];
  #pragma unroll
  for(int r=0; r<8; ++r) creg[r] = 0.f;
  __syncthreads();

  for(int t=0; t<S_; ++t){
    v8f acc0 = {}, acc1 = {}, acc2 = {}, acc3 = {};
    #pragma unroll
    for(int kk=0; kk<KK_HH; ++kk){
      const _Float16* ar = hbuf + col*H_ + kk*32;
      v8h lo = *(const v8h*)(ar +      8*half);
      v8h hi = *(const v8h*)(ar + 16 + 8*half);
      v16h a;
      #pragma unroll
      for(int e=0; e<8; ++e){ a[e] = lo[e]; a[e+8] = hi[e]; }
      v16h b0 = *(const v16h*)(packd + ((size_t)(( 0+wv)*KK_HH + kk)*32 + lane)*16);
      acc0 = __builtin_amdgcn_wmma_f32_16x16x32_f16(false,a,false,b0,(short)0,acc0,false,false);
      v16h b1 = *(const v16h*)(packd + ((size_t)((16+wv)*KK_HH + kk)*32 + lane)*16);
      acc1 = __builtin_amdgcn_wmma_f32_16x16x32_f16(false,a,false,b1,(short)0,acc1,false,false);
      v16h b2 = *(const v16h*)(packd + ((size_t)((32+wv)*KK_HH + kk)*32 + lane)*16);
      acc2 = __builtin_amdgcn_wmma_f32_16x16x32_f16(false,a,false,b2,(short)0,acc2,false,false);
      v16h b3 = *(const v16h*)(packd + ((size_t)((48+wv)*KK_HH + kk)*32 + lane)*16);
      acc3 = __builtin_amdgcn_wmma_f32_16x16x32_f16(false,a,false,b3,(short)0,acc3,false,false);
    }
    __syncthreads();                       // all reads of hbuf done before updates
    #pragma unroll
    for(int r=0; r<8; ++r){
      int m   = r + 8*half;
      int b   = btile*16 + m;
      int len = lens[r];
      bool valid = t < len;
      int tidx = (dir == 0) ? t : (valid ? (len-1-t) : t);
      size_t base = ((size_t)b*S_ + tidx)*G4H;
      float gi = acc0[r] + xgd[base +        u];
      float gf = acc1[r] + xgd[base + 256  + u];
      float gg = acc2[r] + xgd[base + 512  + u];
      float go = acc3[r] + xgd[base + 768  + u];
      float iv = sigmoidf_(gi);
      float fv = sigmoidf_(gf);
      float gv = tanhf(gg);
      float ov = sigmoidf_(go);
      float cn = fv*creg[r] + iv*gv;
      float hn = ov*tanhf(cn);
      int p = (dir == 0) ? t : tidx;       // bwd: write already-re-reversed position
      hcat[((size_t)b*S_ + p)*(2*H_) + dir*H_ + u] = valid ? hn : 0.f;
      if(valid){
        creg[r] = cn;
        hbuf[m*H_ + u] = (_Float16)hn;
      }
    }
    __syncthreads();
  }
}

// ---------------------------------------------------------- emission projection
__global__ void k_proj(const float* __restrict__ hcat, const float* __restrict__ proj_w,
                       const float* __restrict__ proj_b, float* __restrict__ em){
  int tok  = blockIdx.x;
  int lane = threadIdx.x;
  if(lane >= T_) return;
  const float* hrow = hcat + (size_t)tok*2*H_;
  const float* w    = proj_w + lane*2*H_;
  float acc = proj_b[lane];
  for(int j=0; j<2*H_; ++j) acc += hrow[j]*w[j];
  em[(size_t)tok*T_ + lane] = acc;
}

// ------------------------------------------------------------------- CRF NLL
__global__ void k_crf(const float* __restrict__ em, const int* __restrict__ tags,
                      const int* __restrict__ lengths, const float* __restrict__ trans,
                      const float* __restrict__ start_trans, const float* __restrict__ end_trans,
                      float* __restrict__ out){
  int b    = blockIdx.x;
  int lane = threadIdx.x;
  __shared__ float tr[T_*T_];
  __shared__ float alpha[T_];
  for(int i=lane; i<T_*T_; i+=32) tr[i] = trans[i];
  const float* emb = em  + (size_t)b*S_*T_;
  const int*   tgb = tags + b*S_;
  int len = lengths[b];
  if(lane < T_) alpha[lane] = start_trans[lane] + emb[lane];
  __syncthreads();

  float score = 0.f;
  if(lane == 0){
    score = start_trans[tgb[0]] + end_trans[tgb[len-1]];
    for(int t=0; t<len; ++t)  score += emb[t*T_ + tgb[t]];
    for(int t=1; t<len; ++t)  score += tr[tgb[t-1]*T_ + tgb[t]];
  }

  for(int t=1; t<S_; ++t){
    float nxt = 0.f;
    bool act = (lane < T_) && (t < len);
    if(lane < T_){
      float mx = -1e30f;
      for(int i=0; i<T_; ++i) mx = fmaxf(mx, alpha[i] + tr[i*T_ + lane]);
      float s = 0.f;
      for(int i=0; i<T_; ++i) s += __expf(alpha[i] + tr[i*T_ + lane] - mx);
      nxt = mx + __logf(s) + emb[t*T_ + lane];
    }
    __syncthreads();
    if(act) alpha[lane] = nxt;
    __syncthreads();
  }

  if(lane == 0){
    float mx = -1e30f;
    for(int j=0; j<T_; ++j) mx = fmaxf(mx, alpha[j] + end_trans[j]);
    float s = 0.f;
    for(int j=0; j<T_; ++j) s += __expf(alpha[j] + end_trans[j] - mx);
    float logZ = mx + __logf(s);
    atomicAdd(out, logZ - score);
  }
}

// ---------------------------------------------------------------------------
extern "C" void kernel_launch(void* const* d_in, const int* in_sizes, int n_in,
                              void* d_out, int out_size, void* d_ws, size_t ws_size,
                              hipStream_t stream) {
  (void)in_sizes; (void)n_in; (void)out_size; (void)ws_size;
  const int*   words       = (const int*)  d_in[0];
  const int*   chars       = (const int*)  d_in[1];
  const int*   tags        = (const int*)  d_in[2];
  const int*   lengths     = (const int*)  d_in[3];
  const float* word_emb    = (const float*)d_in[4];
  const float* char_emb    = (const float*)d_in[5];
  const float* conv_w      = (const float*)d_in[6];
  const float* conv_b      = (const float*)d_in[7];
  const float* w_ih_f      = (const float*)d_in[8];
  const float* w_hh_f      = (const float*)d_in[9];
  const float* b_f         = (const float*)d_in[10];
  const float* w_ih_b      = (const float*)d_in[11];
  const float* w_hh_b      = (const float*)d_in[12];
  const float* b_b         = (const float*)d_in[13];
  const float* proj_w      = (const float*)d_in[14];
  const float* proj_b      = (const float*)d_in[15];
  const float* trans       = (const float*)d_in[16];
  const float* start_trans = (const float*)d_in[17];
  const float* end_trans   = (const float*)d_in[18];
  float* out = (float*)d_out;

  char* ws = (char*)d_ws;
  size_t off = 0;
  auto alloc = [&](size_t bytes)->void*{
    void* p = ws + off;
    off = (off + bytes + 255) & ~(size_t)255;
    return p;
  };
  _Float16* inp     = (_Float16*)alloc((size_t)NTOK*DP_*sizeof(_Float16));
  _Float16* pack_ih = (_Float16*)alloc((size_t)2*NT_IH*KK_IH*512*sizeof(_Float16));
  _Float16* pack_hh = (_Float16*)alloc((size_t)2*NT_HH*KK_HH*512*sizeof(_Float16));
  float*    xg      = (float*)   alloc((size_t)2*NTOK*G4H*sizeof(float));
  float*    hcat    = (float*)   alloc((size_t)NTOK*2*H_*sizeof(float));
  float*    emis    = (float*)   alloc((size_t)NTOK*T_*sizeof(float));

  k_zero_out<<<1, 32, 0, stream>>>(out);
  k_pack_ih <<<(2*NT_IH*KK_IH*512 + 255)/256, 256, 0, stream>>>(w_ih_f, w_ih_b, pack_ih);
  k_pack_hh <<<(2*NT_HH*KK_HH*512 + 255)/256, 256, 0, stream>>>(w_hh_f, w_hh_b, pack_hh);
  k_embed   <<<NTOK, 128, 0, stream>>>(words, chars, word_emb, char_emb, conv_w, conv_b, inp);
  k_xg      <<<NTOK, 256, 0, stream>>>(inp, pack_ih, b_f, b_b, xg);
  dim3 glstm(8, 2);
  k_lstm    <<<glstm, 512, 0, stream>>>(pack_hh, xg, lengths, hcat);
  k_proj    <<<NTOK, 32, 0, stream>>>(hcat, proj_w, proj_b, emis);
  k_crf     <<<B_, 32, 0, stream>>>(emis, tags, lengths, trans, start_trans, end_trans, out);
}
